// LinearAttention_14104672600568
// MI455X (gfx1250) — matrix operand order repaired and checked by reading, not compiled, not run on hardware
//
#include <hip/hip_runtime.h>
#include <hip/hip_bf16.h>

// ---------------------------------------------------------------------------
// MI455X (gfx1250) fused linear-attention block.
// Heavy GEMMs (QKV projection 384x256x4096/batch and Wo 256x128x4096/batch)
// run through v_wmma_f32_16x16x32_bf16. Softmax / LayerNorm stay fp32.
// ---------------------------------------------------------------------------

typedef __attribute__((ext_vector_type(8)))  float  v8f;
typedef __attribute__((ext_vector_type(16))) __bf16 v16bf;
typedef __attribute__((ext_vector_type(8)))  __bf16 v8bf;

#define DIM      256
#define HIDDEN   128
#define NTOK     4096
#define NB       16

__device__ __forceinline__ v8f wmma_bf16(v16bf a, v16bf b, v8f c) {
  // (neg_a, A, neg_b, B, c_mod, C, reuse_a, reuse_b)
  return __builtin_amdgcn_wmma_f32_16x16x32_bf16(false, a, false, b, (short)0, c,
                                                 false, false);
}

// Build a v16bf A/B fragment from two aligned 16B chunks.
__device__ __forceinline__ v16bf make_v16(const __bf16* p) {
  v8bf lo = *(const v8bf*)(p);
  v8bf hi = *(const v8bf*)(p + 8);
  v16bf r;
#pragma unroll
  for (int i = 0; i < 8; ++i) { r[i] = lo[i]; r[i + 8] = hi[i]; }
  return r;
}

// ---------------------------------------------------------------------------
// Kernel P: pack weights into WMMA A-fragment order (bf16).
// A fragment k order used here (matches B): lane<16 holds row M=lane,
// K = 0..15 contiguous; lane>=16 holds row M=lane-16, K = 16..31 contiguous.
// wf : 24 m-tiles x 8 k-chunks  (Wq rows 0..127, Wk, Wv stacked -> M=384, K=256)
// wof: 16 m-tiles x 4 k-chunks  (Wo: M=256, K=128)
// ---------------------------------------------------------------------------
__global__ void __launch_bounds__(32) pack_weights(
    const float* __restrict__ Wq, const float* __restrict__ Wk,
    const float* __restrict__ Wv, const float* __restrict__ Wo,
    __bf16* __restrict__ wf, __bf16* __restrict__ wof) {
  int bb = blockIdx.x;
  int l = threadIdx.x;
  int r = l & 15, half = l >> 4;
  if (bb < 192) {               // Wq/Wk/Wv fragments
    int mt = bb >> 3, kc = bb & 7;
    const float* W = (mt < 8) ? Wq : ((mt < 16) ? Wk : Wv);
    int row = (mt & 7) * 16 + r;
    int c0 = kc * 32 + half * 16;
    __bf16* dst = wf + ((size_t)bb * 32 + l) * 16;
#pragma unroll
    for (int j = 0; j < 8; ++j) {
      dst[j]     = (__bf16)W[row * DIM + c0 + j];
      dst[8 + j] = (__bf16)W[row * DIM + c0 + 8 + j];
    }
  } else {                      // Wo fragments
    int bb2 = bb - 192;
    int mt = bb2 >> 2, kc = bb2 & 3;
    int row = mt * 16 + r;
    int c0 = kc * 32 + half * 16;
    __bf16* dst = wof + ((size_t)bb2 * 32 + l) * 16;
#pragma unroll
    for (int j = 0; j < 8; ++j) {
      dst[j]     = (__bf16)Wo[row * HIDDEN + c0 + j];
      dst[8 + j] = (__bf16)Wo[row * HIDDEN + c0 + 8 + j];
    }
  }
}

// ---------------------------------------------------------------------------
// Kernel A: QKV projection. Grid (ntiles=64, b=16), 256 threads (8 waves).
// Each workgroup: 64-token tile; x tile staged to LDS as bf16 [token][chan].
// B fragment k order used here (matches A): lane<16 = col N=lane with K=0..15 contiguous;
// lane>=16 = col N=lane-16 with K=16..31 contiguous -> two ds_load_b128.
// Wave w computes m-tiles {w, w+8, w+16} (Q, K, V) over 4 n-tiles.
// ---------------------------------------------------------------------------
__global__ void __launch_bounds__(256) qkv_kernel(
    const float* __restrict__ x, const __bf16* __restrict__ wf,
    float* __restrict__ qkv /* [3][16][128][4096] */) {
  __shared__ __bf16 xs[64][264];            // pitch 264 avoids bank conflicts
  int b = blockIdx.y;
  int n0 = blockIdx.x * 64;
  int t = threadIdx.x;

  { // stage x tile, fp32 -> bf16, transpose to [token][channel]
    int nn4 = (t & 15) * 4;
    int c0 = t >> 4;                        // 0..15
    const float* xb = x + (size_t)b * DIM * NTOK + n0;
#pragma unroll
    for (int k = 0; k < 16; ++k) {
      int c = c0 + 16 * k;
      const float4 v = *(const float4*)(xb + (size_t)c * NTOK + nn4);
      xs[nn4 + 0][c] = (__bf16)v.x;
      xs[nn4 + 1][c] = (__bf16)v.y;
      xs[nn4 + 2][c] = (__bf16)v.z;
      xs[nn4 + 3][c] = (__bf16)v.w;
    }
  }
  __syncthreads();

  int wv = t >> 5, l = t & 31;
  int lo16 = l & 15, hi = l >> 4;

  v8f acc[3][4];
#pragma unroll
  for (int m3 = 0; m3 < 3; ++m3)
#pragma unroll
    for (int nt = 0; nt < 4; ++nt) {
      v8f z = {0.f, 0.f, 0.f, 0.f, 0.f, 0.f, 0.f, 0.f};
      acc[m3][nt] = z;
    }

  for (int kc = 0; kc < 8; ++kc) {
    v16bf afr[3];
#pragma unroll
    for (int m3 = 0; m3 < 3; ++m3) {
      int mt = wv + 8 * m3;
      afr[m3] = make_v16(wf + (((size_t)mt * 8 + kc) * 32 + l) * 16);
    }
#pragma unroll
    for (int nt = 0; nt < 4; ++nt) {
      int nn = nt * 16 + lo16;
      v16bf bfr = make_v16(&xs[nn][kc * 32 + hi * 16]);
#pragma unroll
      for (int m3 = 0; m3 < 3; ++m3)
        acc[m3][nt] = wmma_bf16(afr[m3], bfr, acc[m3][nt]);
    }
  }

  // C/D 32-bit layout: VGPR r -> M = r + (lane>=16 ? 8 : 0), N = lane&15
  int rowbase = wv * 16 + hi * 8;
#pragma unroll
  for (int m3 = 0; m3 < 3; ++m3) {
    float* mb = qkv + (size_t)m3 * (NB * HIDDEN * NTOK) +
                (size_t)b * HIDDEN * NTOK;
#pragma unroll
    for (int nt = 0; nt < 4; ++nt) {
      int col = n0 + nt * 16 + lo16;
#pragma unroll
      for (int r = 0; r < 8; ++r)
        mb[(size_t)(rowbase + r) * NTOK + col] = acc[m3][nt][r];
    }
  }
}

// ---------------------------------------------------------------------------
// Kernel B: per-row max & sum(exp) of K logits (softmax over tokens).
// Grid 2048 (= b*128), 256 threads.
// ---------------------------------------------------------------------------
__global__ void __launch_bounds__(256) krowstats_kernel(
    const float* __restrict__ kws, float* __restrict__ kmax,
    float* __restrict__ ksum) {
  __shared__ float red[256];
  int row = blockIdx.x;
  const float* p = kws + (size_t)row * NTOK;
  int t = threadIdx.x;
  float m = -3.402823466e38f;
  for (int i = t; i < NTOK; i += 256) m = fmaxf(m, p[i]);
  red[t] = m;
  __syncthreads();
  for (int s = 128; s > 0; s >>= 1) {
    if (t < s) red[t] = fmaxf(red[t], red[t + s]);
    __syncthreads();
  }
  float rowmax = red[0];
  __syncthreads();
  float s = 0.f;
  for (int i = t; i < NTOK; i += 256) s += __expf(p[i] - rowmax);
  red[t] = s;
  __syncthreads();
  for (int st = 128; st > 0; st >>= 1) {
    if (t < st) red[t] += red[t + st];
    __syncthreads();
  }
  if (t == 0) { kmax[row] = rowmax; ksum[row] = red[0]; }
}

// ---------------------------------------------------------------------------
// Kernel C: context[b,h,d,e] = sum_n softmax(K)[d,n] * (V[e,n]/n).
// Grid (h=4, b=16), 256 threads. LDS-chunked, exp folded in at load.
// ---------------------------------------------------------------------------
__global__ void __launch_bounds__(256) context_kernel(
    const float* __restrict__ kws, const float* __restrict__ vws,
    const float* __restrict__ kmax, const float* __restrict__ ksum,
    float* __restrict__ ctx /* [16][4][32][32] */) {
  __shared__ float ks[32][132];
  __shared__ float vs[32][132];
  int h = blockIdx.x, b = blockIdx.y, t = threadIdx.x;
  const float* kb = kws + ((size_t)b * HIDDEN + h * 32) * NTOK;
  const float* vb = vws + ((size_t)b * HIDDEN + h * 32) * NTOK;
  int dl = t >> 3;                 // loader & compute row 0..31
  int j0 = (t & 7) * 16;
  float km  = kmax[b * HIDDEN + h * 32 + dl];
  float inv = (1.0f / ksum[b * HIDDEN + h * 32 + dl]) * (1.0f / (float)NTOK);
  int e0 = (t & 7) * 4;            // 4 consecutive e per thread
  float a0 = 0.f, a1 = 0.f, a2 = 0.f, a3 = 0.f;
  for (int c0 = 0; c0 < NTOK; c0 += 128) {
    __syncthreads();
#pragma unroll
    for (int j = 0; j < 16; ++j) {
      ks[dl][j0 + j] = __expf(kb[(size_t)dl * NTOK + c0 + j0 + j] - km) * inv;
      vs[dl][j0 + j] = vb[(size_t)dl * NTOK + c0 + j0 + j];
    }
    __syncthreads();
#pragma unroll 4
    for (int nn = 0; nn < 128; ++nn) {
      float w = ks[dl][nn];
      a0 += w * vs[e0 + 0][nn];
      a1 += w * vs[e0 + 1][nn];
      a2 += w * vs[e0 + 2][nn];
      a3 += w * vs[e0 + 3][nn];
    }
  }
  float* cp = ctx + (((size_t)b * 4 + h) * 32 + dl) * 32 + e0;
  cp[0] = a0; cp[1] = a1; cp[2] = a2; cp[3] = a3;
}

// ---------------------------------------------------------------------------
// Kernel D: Q head-dim softmax -> ctx^T x Q -> Wo GEMM (WMMA) + bo -> LayerNorm.
// Grid (ntiles=128, b=16), 256 threads, 32-token tile. LDS < 64KB via aliasing.
// ---------------------------------------------------------------------------
__global__ void __launch_bounds__(256) attn_out_kernel(
    const float* __restrict__ qws, const float* __restrict__ ctx,
    const __bf16* __restrict__ wof, const float* __restrict__ bo,
    const float* __restrict__ g, float* __restrict__ out) {
  __shared__ float smem0[256 * 36];          // aliases: qs[32][132] -> outs[256][36]
  __shared__ __bf16 osb[32][136];            // bf16 B-operand for Wo GEMM
  __shared__ float gs[256], bos[256];
  __shared__ float mean_s[32], rstd_s[32];
  float (*qs)[132]  = (float (*)[132])smem0;
  float (*outs)[36] = (float (*)[36])smem0;

  int b = blockIdx.y, n0 = blockIdx.x * 32, t = threadIdx.x;
  gs[t]  = g[t];
  bos[t] = bo[t];

  { // stage Q tile [token][channel]
    int nn = t & 31, r0 = t >> 5;
    const float* qb = qws + (size_t)b * HIDDEN * NTOK + n0;
#pragma unroll
    for (int k = 0; k < 16; ++k) {
      int d = r0 + 8 * k;
      qs[nn][d] = qb[(size_t)d * NTOK + nn];
    }
  }
  __syncthreads();

  // softmax over head-dim (32) per (token, head), * 32^-0.5
  if (t < 128) {
    int nn = t >> 2, h = t & 3;
    float* p = &qs[nn][h * 32];
    float m = -3.402823466e38f;
#pragma unroll
    for (int e = 0; e < 32; ++e) m = fmaxf(m, p[e]);
    float s = 0.f;
#pragma unroll
    for (int e = 0; e < 32; ++e) { float w = __expf(p[e] - m); p[e] = w; s += w; }
    float inv = 0.1767766952966369f / s;     // scale / sum
#pragma unroll
    for (int e = 0; e < 32; ++e) p[e] *= inv;
  }
  __syncthreads();

  { // out_h[e,n] = sum_d ctx[d,e] * q[d,n]  (VALU; tiny), store bf16 to osb
    int e = t & 127, h = e >> 5, eh = e & 31;
    int nb = (t >> 7) * 16;
    const float* cb = ctx + (((size_t)b * 4 + h) * 32) * 32 + eh;
    float a[16];
#pragma unroll
    for (int j = 0; j < 16; ++j) a[j] = 0.f;
    for (int d = 0; d < 32; ++d) {
      float cv = cb[d * 32];
#pragma unroll
      for (int j = 0; j < 16; ++j) a[j] += cv * qs[nb + j][h * 32 + d];
    }
#pragma unroll
    for (int j = 0; j < 16; ++j) osb[nb + j][e] = (__bf16)a[j];
  }
  __syncthreads();

  // Wo GEMM: M=256, K=128, N=32 via WMMA bf16
  int wv = t >> 5, l = t & 31, lo16 = l & 15, hi = l >> 4;
  v8f acc2[2][2];
#pragma unroll
  for (int m2 = 0; m2 < 2; ++m2)
#pragma unroll
    for (int nt = 0; nt < 2; ++nt) {
      v8f z = {0.f, 0.f, 0.f, 0.f, 0.f, 0.f, 0.f, 0.f};
      acc2[m2][nt] = z;
    }
  for (int kc = 0; kc < 4; ++kc) {
    v16bf afr[2];
#pragma unroll
    for (int m2 = 0; m2 < 2; ++m2) {
      int mt = wv + 8 * m2;
      afr[m2] = make_v16(wof + (((size_t)mt * 4 + kc) * 32 + l) * 16);
    }
#pragma unroll
    for (int nt = 0; nt < 2; ++nt) {
      v16bf bfr = make_v16(&osb[nt * 16 + lo16][kc * 32 + hi * 16]);
#pragma unroll
      for (int m2 = 0; m2 < 2; ++m2)
        acc2[m2][nt] = wmma_bf16(afr[m2], bfr, acc2[m2][nt]);
    }
  }
  // + bo, stash fp32 in LDS (aliases qs; all qs reads are behind a barrier)
#pragma unroll
  for (int m2 = 0; m2 < 2; ++m2) {
    int cb0 = (wv + 8 * m2) * 16 + hi * 8;
#pragma unroll
    for (int nt = 0; nt < 2; ++nt) {
      int nn = nt * 16 + lo16;
#pragma unroll
      for (int r = 0; r < 8; ++r) {
        int c = cb0 + r;
        outs[c][nn] = acc2[m2][nt][r] + bos[c];
      }
    }
  }
  __syncthreads();

  // LayerNorm stats per token (biased var, eps 1e-5)
  if (t < 32) {
    float s = 0.f, ss = 0.f;
    for (int c = 0; c < 256; ++c) { float v = outs[c][t]; s += v; ss += v * v; }
    float mean = s * (1.0f / 256.0f);
    float var = ss * (1.0f / 256.0f) - mean * mean;
    mean_s[t] = mean;
    rstd_s[t] = rsqrtf(var + 1e-5f);
  }
  __syncthreads();

  { // normalized, *g, coalesced store
    int nn = t & 31, c0 = t >> 5;
    float mu = mean_s[nn], rs = rstd_s[nn];
    float* ob = out + (size_t)b * DIM * NTOK + n0;
#pragma unroll
    for (int k = 0; k < 32; ++k) {
      int c = c0 + 8 * k;
      ob[(size_t)c * NTOK + nn] = (outs[c][nn] - mu) * rs * gs[c];
    }
  }
}

// ---------------------------------------------------------------------------
extern "C" void kernel_launch(void* const* d_in, const int* in_sizes, int n_in,
                              void* d_out, int out_size, void* d_ws, size_t ws_size,
                              hipStream_t stream) {
  (void)in_sizes; (void)n_in; (void)out_size; (void)ws_size;
  const float* x  = (const float*)d_in[0];
  const float* Wq = (const float*)d_in[1];
  const float* Wk = (const float*)d_in[2];
  const float* Wv = (const float*)d_in[3];
  const float* Wo = (const float*)d_in[4];
  const float* bo = (const float*)d_in[5];
  const float* g  = (const float*)d_in[6];
  float* out = (float*)d_out;

  char* ws = (char*)d_ws;
  float* qkv  = (float*)ws;                                // [3][16][128][4096] fp32
  float* qw   = qkv;
  float* kw   = qkv + (size_t)NB * HIDDEN * NTOK;
  float* vw   = qkv + (size_t)2 * NB * HIDDEN * NTOK;
  size_t off  = (size_t)3 * NB * HIDDEN * NTOK * sizeof(float);
  float* kmax = (float*)(ws + off);                        // [2048]
  float* ksum = kmax + 2048;                               // [2048]
  float* ctx  = ksum + 2048;                               // [16][4][32][32]
  __bf16* wf  = (__bf16*)(ctx + 16 * 4 * 32 * 32);         // 24*8*32*16 bf16
  __bf16* wof = wf + 24 * 8 * 32 * 16;                     // 16*4*32*16 bf16

  pack_weights<<<dim3(224), dim3(32), 0, stream>>>(Wq, Wk, Wv, Wo, wf, wof);
  qkv_kernel<<<dim3(NTOK / 64, NB), dim3(256), 0, stream>>>(x, wf, qkv);
  krowstats_kernel<<<dim3(NB * HIDDEN), dim3(256), 0, stream>>>(kw, kmax, ksum);
  context_kernel<<<dim3(4, NB), dim3(256), 0, stream>>>(kw, vw, kmax, ksum, ctx);
  attn_out_kernel<<<dim3(NTOK / 32, NB), dim3(256), 0, stream>>>(qw, ctx, wof, bo,
                                                                 g, out);
}
